// WFNet_51994874085406
// MI455X (gfx1250) — matrix operand order repaired and checked by reading, not compiled, not run on hardware
//
#include <hip/hip_runtime.h>

typedef __attribute__((ext_vector_type(16))) _Float16 v16h;
typedef __attribute__((ext_vector_type(8)))  float    v8f;

#define N_ELX      16
#define N_NUCX     4
#define N_PAIRSX   184      // 16*4 + 120
#define HIDDENX    64
#define DIST_STRIDE 185     // odd stride -> conflict-free LDS column reads
#define H_STRIDE    65      // 64+1 pad
#define B1_ELEMS   (184*4*32*16)   // K-chunks * N-tiles * lanes * f16-per-lane
#define B2_ELEMS   (2*4*32*16)

__device__ __forceinline__ float sspf(float x) {
  // shifted softplus, overflow-safe: max(x,0) + log(1+exp(-|x|)) - log(2)
  float ax = fabsf(x);
  return fmaxf(x, 0.0f) + __logf(1.0f + __expf(-ax)) - 0.69314718056f;
}

// ---------------------------------------------------------------------------
// Repack W1 (f32 [5888,64]) into WMMA B-fragment order, f16.
// B 32x16 f16 fragment packing used here (contiguous k, same as A): lane column n = lane&15,
// K = (lane>>4)*16 + slot, slot = 0..15 (2 per VGPR).
// Dest index = ((p*4 + t)*32 + lane)*16 + slot  == linear tid (coalesced store)
// ---------------------------------------------------------------------------
__global__ void wfnet_repack_w1(const float* __restrict__ W1,
                                _Float16* __restrict__ dst) {
  int idx = blockIdx.x * blockDim.x + threadIdx.x;
  if (idx >= B1_ELEMS) return;
  int s    = idx & 15;
  int lane = (idx >> 4) & 31;
  int t    = (idx >> 9) & 3;
  int p    = idx >> 11;                 // K-chunk (= pair index), 0..183
  int k    = p * 32 + (lane >> 4) * 16 + s;
  int n    = t * 16 + (lane & 15);
  dst[idx] = (_Float16)W1[k * HIDDENX + n];
}

__global__ void wfnet_repack_w2(const float* __restrict__ W2,
                                _Float16* __restrict__ dst) {
  int idx = blockIdx.x * blockDim.x + threadIdx.x;
  if (idx >= B2_ELEMS) return;
  int s    = idx & 15;
  int lane = (idx >> 4) & 31;
  int t    = (idx >> 9) & 3;
  int p    = idx >> 11;                 // 0..1
  int k    = p * 32 + (lane >> 4) * 16 + s;
  int n    = t * 16 + (lane & 15);
  dst[idx] = (_Float16)W2[k * HIDDENX + n];
}

// ---------------------------------------------------------------------------
// Fused: distances -> Gaussian basis (on the fly, f16) -> WMMA GEMM1 (K=5888)
//        -> ssp -> WMMA GEMM2 (K=64) -> ssp -> dot W3 -> exp(y)*asymp
// Block: 128 threads (4 waves). Each wave owns 16 batch rows, all 64 hidden
// columns (4 N-tiles of 16), loops 184 K-chunks of 32.
// ---------------------------------------------------------------------------
__global__ __launch_bounds__(128) void wfnet_main(
    const float* __restrict__ rs, const float* __restrict__ coords,
    const float* __restrict__ charges,
    const float* __restrict__ bias1, const float* __restrict__ bias2,
    const float* __restrict__ W3, const float* __restrict__ b3,
    const _Float16* __restrict__ B1, const _Float16* __restrict__ B2,
    float* __restrict__ out) {

  __shared__ float lds_dist[64 * DIST_STRIDE];   // 64 rows x 184 dists (padded)
  __shared__ float lds_h[4 * N_ELX * H_STRIDE];  // per-wave 16x64 activations

  const int tid  = threadIdx.x;
  const int lane = tid & 31;
  const int wv   = tid >> 5;
  const int hh   = lane >> 4;          // lane half
  const int mrow = lane & 15;          // A-matrix row / C column index
  const int base = blockIdx.x * 64;    // first batch row of this block

  // ---- Phase 1: all 64*184 distances into LDS (cooperative) ----------------
  for (int task = tid; task < 64 * N_PAIRSX; task += 128) {
    int rl  = task / N_PAIRSX;
    int pp  = task - rl * N_PAIRSX;
    int row = base + rl;
    float d;
    if (pp < N_ELX * N_NUCX) {
      int e  = pp >> 2;
      int nu = pp & 3;
      float dx = rs[(row * N_ELX + e) * 3 + 0] - coords[nu * 3 + 0];
      float dy = rs[(row * N_ELX + e) * 3 + 1] - coords[nu * 3 + 1];
      float dz = rs[(row * N_ELX + e) * 3 + 2] - coords[nu * 3 + 2];
      d = sqrtf(dx * dx + dy * dy + dz * dz);
    } else {
      int q = pp - N_ELX * N_NUCX;
      int i = 0, rem = q;
      for (int ii = 0; ii < N_ELX - 1; ++ii) {
        int cnt = N_ELX - 1 - ii;
        if (rem < cnt) { i = ii; break; }
        rem -= cnt;
      }
      int j = i + 1 + rem;
      float dx = rs[(row * N_ELX + i) * 3 + 0] - rs[(row * N_ELX + j) * 3 + 0];
      float dy = rs[(row * N_ELX + i) * 3 + 1] - rs[(row * N_ELX + j) * 3 + 1];
      float dz = rs[(row * N_ELX + i) * 3 + 2] - rs[(row * N_ELX + j) * 3 + 2];
      d = sqrtf(dx * dx + dy * dy + dz * dz);
    }
    lds_dist[rl * DIST_STRIDE + pp] = d;
  }
  __syncthreads();

  // ---- Per-lane basis constants for this lane's 16 A-fragment slots --------
  // A 16x32 f16 layout (matches B packing): slot s -> K = (lane>>4)*16 + s
  // K(s) = hh*16 + s
  float muv[16], is2v[16];
  int   kidx[16];
#pragma unroll
  for (int s = 0; s < 16; ++s) {
    int K = hh * 16 + s;
    kidx[s] = K;
    float q  = (float)K * (1.0f / 31.0f);
    float mu = 10.0f * q * q;                  // CUTOFF * q^2
    float sg = (1.0f + 10.0f * q) * (1.0f / 7.0f);
    muv[s]  = mu;
    is2v[s] = 1.0f / (sg * sg);
  }

  // ---- GEMM1: 184 K-chunks, 4 N-tiles, fused feature generation ------------
  v8f acc0 = {}, acc1 = {}, acc2 = {}, acc3 = {};
  const float* my_dist = &lds_dist[(wv * 16 + mrow) * DIST_STRIDE];
  const v16h* Bbase = (const v16h*)B1;

  for (int p = 0; p < N_PAIRSX; ++p) {
    float dd = my_dist[p];
    v16h a;
#pragma unroll
    for (int s = 0; s < 16; ++s) {
      float t = dd - muv[s];
      a[s] = (_Float16)__expf(-t * t * is2v[s]);
    }
    const v16h* bp = Bbase + p * 128 + lane;   // ((p*4+t)*32+lane)
    v16h bf0 = bp[0];
    v16h bf1 = bp[32];
    v16h bf2 = bp[64];
    v16h bf3 = bp[96];
    acc0 = __builtin_amdgcn_wmma_f32_16x16x32_f16(false, a, false, bf0,
                                                  (short)0, acc0, false, false);
    acc1 = __builtin_amdgcn_wmma_f32_16x16x32_f16(false, a, false, bf1,
                                                  (short)0, acc1, false, false);
    acc2 = __builtin_amdgcn_wmma_f32_16x16x32_f16(false, a, false, bf2,
                                                  (short)0, acc2, false, false);
    acc3 = __builtin_amdgcn_wmma_f32_16x16x32_f16(false, a, false, bf3,
                                                  (short)0, acc3, false, false);
  }

  // ---- bias + ssp, stash h1 in LDS (per-wave region, stride 65) ------------
  float* hbuf = &lds_h[wv * N_ELX * H_STRIDE];
  {
    v8f accs[4] = {acc0, acc1, acc2, acc3};
#pragma unroll
    for (int t = 0; t < 4; ++t) {
      int n = t * 16 + mrow;
      float bv = bias1[n];
#pragma unroll
      for (int r = 0; r < 8; ++r) {
        int m = r + 8 * hh;
        hbuf[m * H_STRIDE + n] = sspf(accs[t][r] + bv);
      }
    }
  }

  // ---- GEMM2: h1[16x64] x W2[64x64], 2 K-chunks x 4 N-tiles ----------------
  v8f g0 = {}, g1 = {}, g2 = {}, g3 = {};
  const v16h* B2base = (const v16h*)B2;
#pragma unroll
  for (int p2 = 0; p2 < 2; ++p2) {
    v16h a2;
#pragma unroll
    for (int s = 0; s < 16; ++s)
      a2[s] = (_Float16)hbuf[mrow * H_STRIDE + 32 * p2 + kidx[s]];
    const v16h* bp = B2base + p2 * 128 + lane;
    v16h bf0 = bp[0];
    v16h bf1 = bp[32];
    v16h bf2 = bp[64];
    v16h bf3 = bp[96];
    g0 = __builtin_amdgcn_wmma_f32_16x16x32_f16(false, a2, false, bf0,
                                                (short)0, g0, false, false);
    g1 = __builtin_amdgcn_wmma_f32_16x16x32_f16(false, a2, false, bf1,
                                                (short)0, g1, false, false);
    g2 = __builtin_amdgcn_wmma_f32_16x16x32_f16(false, a2, false, bf2,
                                                (short)0, g2, false, false);
    g3 = __builtin_amdgcn_wmma_f32_16x16x32_f16(false, a2, false, bf3,
                                                (short)0, g3, false, false);
  }

  // ---- bias + ssp -> h2 back into LDS (overwrites h1, same wave, in-order) -
  {
    v8f accs[4] = {g0, g1, g2, g3};
#pragma unroll
    for (int t = 0; t < 4; ++t) {
      int n = t * 16 + mrow;
      float bv = bias2[n];
#pragma unroll
      for (int r = 0; r < 8; ++r) {
        int m = r + 8 * hh;
        hbuf[m * H_STRIDE + n] = sspf(accs[t][r] + bv);
      }
    }
  }

  // ---- Epilogue: y = h2.W3 + b3 ; asymptotic nuclear factor ; output -------
  if (lane < 16) {
    int rl  = wv * 16 + lane;
    int row = base + rl;
    float y = b3[0];
    for (int n = 0; n < HIDDENX; ++n)
      y += hbuf[lane * H_STRIDE + n] * W3[n];
    // decay = sqrt(2*0.5) = 1, ALPHA = 1:
    // term = (Z*d + d^2) / (1 + d)
    float ssum = 0.0f;
    const float* dn = &lds_dist[rl * DIST_STRIDE];
#pragma unroll 4
    for (int e = 0; e < N_ELX; ++e) {
      for (int nu = 0; nu < N_NUCX; ++nu) {
        float d = dn[e * 4 + nu];
        ssum += (charges[nu] * d + d * d) / (1.0f + d);
      }
    }
    out[row] = __expf(y - ssum);
  }
}

extern "C" void kernel_launch(void* const* d_in, const int* in_sizes, int n_in,
                              void* d_out, int out_size, void* d_ws, size_t ws_size,
                              hipStream_t stream) {
  const float* rs      = (const float*)d_in[0];
  const float* coords  = (const float*)d_in[1];
  const float* charges = (const float*)d_in[2];
  const float* W1      = (const float*)d_in[3];
  const float* b1      = (const float*)d_in[4];
  const float* W2      = (const float*)d_in[5];
  const float* b2      = (const float*)d_in[6];
  const float* W3      = (const float*)d_in[7];
  const float* b3      = (const float*)d_in[8];
  float* out = (float*)d_out;

  int B = in_sizes[0] / (N_ELX * 3);   // 32768

  _Float16* B1p = (_Float16*)d_ws;
  _Float16* B2p = B1p + B1_ELEMS;      // 753,664 B + 8,192 B in workspace

  wfnet_repack_w1<<<(B1_ELEMS + 255) / 256, 256, 0, stream>>>(W1, B1p);
  wfnet_repack_w2<<<(B2_ELEMS + 255) / 256, 256, 0, stream>>>(W2, B2p);

  wfnet_main<<<B / 64, 128, 0, stream>>>(rs, coords, charges, b1, b2, W3, b3,
                                         B1p, B2p, out);
}
